// Transformer_876173329025
// MI455X (gfx1250) — matrix operand order repaired and checked by reading, not compiled, not run on hardware
//
#include <hip/hip_runtime.h>

// ---------------------------------------------------------------------------
// CDNA5 (gfx1250) transformer forward. All GEMMs via v_wmma_f32_16x16x32_bf16.
// wave32, 256-thread blocks (8 waves), 128x64x32 workgroup tiles.
// A tiles staged via GLOBAL_LOAD_ASYNC_TO_LDS_B128 when available; A and B
// fragments both read as contiguous 2x ds_load_b128 per fragment.
// ---------------------------------------------------------------------------

typedef __attribute__((ext_vector_type(16))) __bf16 v16bf;
typedef __attribute__((ext_vector_type(8)))  float  v8f;
typedef int v4i_t __attribute__((vector_size(16)));

#define EMBD 1024
#define SEQL 1024
#define BATCH 2
#define NHEAD 16
#define HDIM 64
#define FFND 4096
#define VOCAB 32000
#define BPAD 40  // padded BsT row length (bf16 elems)

#if defined(__AMDGCN__) && __has_builtin(__builtin_amdgcn_global_load_async_to_lds_b128) && __has_builtin(__builtin_amdgcn_s_wait_asynccnt)
#define HAS_ASYNC_LDS 1
#define AS1G __attribute__((address_space(1)))
#define AS3L __attribute__((address_space(3)))
#else
#define HAS_ASYNC_LDS 0
#endif

__device__ __forceinline__ void wait_async_lds() {
#if HAS_ASYNC_LDS
  __builtin_amdgcn_s_wait_asynccnt(0);
#endif
}

__device__ __forceinline__ v8f wmma_bf16(v16bf a, v16bf b, v8f c) {
  return __builtin_amdgcn_wmma_f32_16x16x32_bf16(
      /*neg_a=*/false, a, /*neg_b=*/false, b,
      /*c_mod=*/(short)0, c, /*reuse_a=*/false, /*reuse_b=*/false);
}

// A fragment: 16x32 bf16 tile at `base` (LDS), row stride `ld` (elems).
// Lanes 0-15: M=lane, elems 0..15 = K0..15.
// Lanes 16-31: M=lane-16, elems 0..15 = K16..31.
__device__ __forceinline__ v16bf frag_a(const __bf16* base, int ld) {
  int lane = threadIdx.x & 31;
  int r = lane & 15, half = lane >> 4;
  const __bf16* p = base + r * ld;
  v16bf a;
#pragma unroll
  for (int j = 0; j < 8; ++j) a[j] = p[half * 16 + j];
#pragma unroll
  for (int j = 0; j < 8; ++j) a[8 + j] = p[half * 16 + 8 + j];
  return a;
}

// B fragment from transposed LDS tile BsT[col][K] (row stride BPAD):
// lane's column = base_col + (lane&15); elems j = K (lane>>4)*16 + j.
// 16 contiguous bf16 per lane -> 2x ds_load_b128.
__device__ __forceinline__ v16bf frag_bT(const __bf16* base) {
  int lane = threadIdx.x & 31;
  int c = lane & 15, half = lane >> 4;
  const __bf16* p = base + c * BPAD + half * 16;
  v16bf b;
#pragma unroll
  for (int j = 0; j < 16; ++j) b[j] = p[j];
  return b;
}

// Stage a 128x32 bf16 tile (row-major, row stride ld) into As[128][32].
__device__ __forceinline__ void stage_a_128x32(const __bf16* __restrict__ G,
                                               size_t ld, __bf16 (*As)[32]) {
  int tid = threadIdx.x;
#if HAS_ASYNC_LDS
#pragma unroll
  for (int i = 0; i < 2; ++i) {
    int e = (tid + i * 256) * 8;  // 8 bf16 = 16 B per lane per issue
    int r = e >> 5, kk = e & 31;
    __builtin_amdgcn_global_load_async_to_lds_b128(
        (AS1G v4i_t*)(G + (size_t)r * ld + kk),
        (AS3L v4i_t*)&As[r][kk], 0, 0);
  }
#else
#pragma unroll
  for (int i = 0; i < 16; ++i) {
    int idx = tid + i * 256;
    int r = idx >> 5, kk = idx & 31;
    As[r][kk] = G[(size_t)r * ld + kk];
  }
#endif
}

// BsT[cc][kk] = G[cc*ld + kk]  (64 rows of 32 contiguous bf16) -- async-able.
__device__ __forceinline__ void stage_bT_rows(const __bf16* __restrict__ G,
                                              size_t ld, __bf16 (*BsT)[BPAD]) {
  int tid = threadIdx.x;
  int cc = tid & 63, g = tid >> 6;  // g in 0..3, 8 elems each
#if HAS_ASYNC_LDS
  __builtin_amdgcn_global_load_async_to_lds_b128(
      (AS1G v4i_t*)(G + (size_t)cc * ld + g * 8),
      (AS3L v4i_t*)&BsT[cc][g * 8], 0, 0);
#else
#pragma unroll
  for (int j = 0; j < 8; ++j) BsT[cc][g * 8 + j] = G[(size_t)cc * ld + g * 8 + j];
#endif
}

// BsT[cc][kk] = G[kk*ld + cc]  (transpose of row-major KxN tile, manual).
__device__ __forceinline__ void stage_bT_transpose(const __bf16* __restrict__ G,
                                                   size_t ld, __bf16 (*BsT)[BPAD]) {
  int tid = threadIdx.x;
#pragma unroll
  for (int i = 0; i < 8; ++i) {
    int idx = tid + i * 256;
    int kk = idx >> 6, cc = idx & 63;
    BsT[cc][kk] = G[(size_t)kk * ld + cc];
  }
}

// C/D layout: lane col = lane&15, VGPR r -> row r + 8*(lane>>4)
template <int EPI>  // bit0: +bias[col], bit1: relu
__device__ __forceinline__ void epilogue_store(float* __restrict__ C,
                                               __bf16* __restrict__ Cbf,
                                               size_t ldc, int gm, int gn,
                                               const v8f& acc,
                                               const float* __restrict__ bias) {
  int lane = threadIdx.x & 31;
  int c = lane & 15, half = lane >> 4;
#pragma unroll
  for (int r = 0; r < 8; ++r) {
    int row = gm + half * 8 + r;
    int col = gn + c;
    float v = acc[r];
    if (EPI & 1) v += bias[col];
    if (EPI & 2) v = fmaxf(v, 0.0f);
    if (C)   C[(size_t)row * ldc + col] = v;
    if (Cbf) Cbf[(size_t)row * ldc + col] = (__bf16)v;
  }
}

// ---------------------------------------------------------------------------
// GEMM: C[MxN] (+bf16 mirror) = A[MxK] bf16 * B[KxN] bf16 (both row-major).
// M%128==0, N%64==0, K%32==0 (all shapes in this model satisfy this).
// ---------------------------------------------------------------------------
template <int EPI>
__global__ __launch_bounds__(256) void gemm_kernel(
    const __bf16* __restrict__ A, const __bf16* __restrict__ B,
    const float* __restrict__ bias, float* __restrict__ C,
    __bf16* __restrict__ Cbf, int M, int N, int K) {
  __shared__ __bf16 As[128][32];
  __shared__ __bf16 BsT[64][BPAD];
  int tid = threadIdx.x;
  int wave = tid >> 5;
  int wm = wave >> 1, wn = wave & 1;  // 4x2 wave grid, each wave 32x32 out
  int bm = blockIdx.y * 128, bn = blockIdx.x * 64;

  v8f acc[2][2] = {};
  for (int k0 = 0; k0 < K; k0 += 32) {
    stage_a_128x32(A + (size_t)bm * K + k0, K, As);
    stage_bT_transpose(B + (size_t)k0 * N + bn, N, BsT);
    if (k0 + 32 < K)
      __builtin_prefetch(B + (size_t)(k0 + 32) * N + bn + (tid & 63), 0, 1);
    wait_async_lds();
    __syncthreads();
    v16bf a0 = frag_a(&As[wm * 32][0], 32);
    v16bf a1 = frag_a(&As[wm * 32 + 16][0], 32);
    v16bf b0 = frag_bT(&BsT[wn * 32][0]);
    v16bf b1 = frag_bT(&BsT[wn * 32 + 16][0]);
    acc[0][0] = wmma_bf16(a0, b0, acc[0][0]);
    acc[0][1] = wmma_bf16(a0, b1, acc[0][1]);
    acc[1][0] = wmma_bf16(a1, b0, acc[1][0]);
    acc[1][1] = wmma_bf16(a1, b1, acc[1][1]);
    __syncthreads();
  }
#pragma unroll
  for (int mi = 0; mi < 2; ++mi)
#pragma unroll
    for (int ni = 0; ni < 2; ++ni)
      epilogue_store<EPI>(C, Cbf, N, bm + wm * 32 + mi * 16,
                          bn + wn * 32 + ni * 16, acc[mi][ni], bias);
}

// ---------------------------------------------------------------------------
// Attention scores: S[bh][q][k] = (Q[b,q,h,:] . K[b,k,h,:]) / 8, causal mask.
// Q,K bf16, stored [B*L, E], head h at column h*64. grid(L/64, L/128, B*H)
// ---------------------------------------------------------------------------
__global__ __launch_bounds__(256) void attn_scores_kernel(
    const __bf16* __restrict__ Q, const __bf16* __restrict__ Km,
    float* __restrict__ S, int causal) {
  int bh = blockIdx.z;
  int b = bh >> 4, h = bh & 15;
  const __bf16* Qb = Q + (size_t)b * SEQL * EMBD + h * HDIM;
  const __bf16* Kb = Km + (size_t)b * SEQL * EMBD + h * HDIM;
  float* Sb = S + (size_t)bh * SEQL * SEQL;

  __shared__ __bf16 As[128][32];
  __shared__ __bf16 BsT[64][BPAD];
  int tid = threadIdx.x;
  int wave = tid >> 5;
  int wm = wave >> 1, wn = wave & 1;
  int bm = blockIdx.y * 128, bn = blockIdx.x * 64;

  v8f acc[2][2] = {};
  for (int k0 = 0; k0 < HDIM; k0 += 32) {
    stage_a_128x32(Qb + (size_t)bm * EMBD + k0, EMBD, As);
    // BsT[key][kdim]: rows of K are already contiguous in kdim -> async path
    stage_bT_rows(Kb + (size_t)bn * EMBD + k0, EMBD, BsT);
    wait_async_lds();
    __syncthreads();
    v16bf a0 = frag_a(&As[wm * 32][0], 32);
    v16bf a1 = frag_a(&As[wm * 32 + 16][0], 32);
    v16bf b0 = frag_bT(&BsT[wn * 32][0]);
    v16bf b1 = frag_bT(&BsT[wn * 32 + 16][0]);
    acc[0][0] = wmma_bf16(a0, b0, acc[0][0]);
    acc[0][1] = wmma_bf16(a0, b1, acc[0][1]);
    acc[1][0] = wmma_bf16(a1, b0, acc[1][0]);
    acc[1][1] = wmma_bf16(a1, b1, acc[1][1]);
    __syncthreads();
  }
  int lane = tid & 31, c = lane & 15, half = lane >> 4;
#pragma unroll
  for (int mi = 0; mi < 2; ++mi)
#pragma unroll
    for (int ni = 0; ni < 2; ++ni)
#pragma unroll
      for (int r = 0; r < 8; ++r) {
        int q = bm + wm * 32 + mi * 16 + half * 8 + r;
        int kp = bn + wn * 32 + ni * 16 + c;
        float v = acc[mi][ni][r] * 0.125f;  // 1/sqrt(64)
        if (causal && kp > q) v = -1e30f;
        Sb[(size_t)q * SEQL + kp] = v;
      }
}

// Row softmax over L=1024 entries, output bf16 probabilities. grid = B*H*L rows
__global__ __launch_bounds__(256) void softmax_kernel(
    const float* __restrict__ S, __bf16* __restrict__ P) {
  size_t row = blockIdx.x;
  const float* s = S + row * SEQL;
  __bf16* p = P + row * SEQL;
  __shared__ float red[256];
  float m = -1e38f;
  for (int i = threadIdx.x; i < SEQL; i += 256) m = fmaxf(m, s[i]);
  red[threadIdx.x] = m;
  __syncthreads();
  for (int off = 128; off > 0; off >>= 1) {
    if (threadIdx.x < off)
      red[threadIdx.x] = fmaxf(red[threadIdx.x], red[threadIdx.x + off]);
    __syncthreads();
  }
  m = red[0];
  __syncthreads();
  float sum = 0.0f;
  for (int i = threadIdx.x; i < SEQL; i += 256) sum += __expf(s[i] - m);
  red[threadIdx.x] = sum;
  __syncthreads();
  for (int off = 128; off > 0; off >>= 1) {
    if (threadIdx.x < off) red[threadIdx.x] += red[threadIdx.x + off];
    __syncthreads();
  }
  float inv = 1.0f / red[0];
  for (int i = threadIdx.x; i < SEQL; i += 256)
    p[i] = (__bf16)(__expf(s[i] - m) * inv);
}

// Obf[b,q,h,:] = sum_k P[bh][q][k] * V[b,k,h,:]. grid(1, L/128, B*H), N=64.
__global__ __launch_bounds__(256) void attn_pv_kernel(
    const __bf16* __restrict__ P, const __bf16* __restrict__ V,
    __bf16* __restrict__ Obf) {
  int bh = blockIdx.z;
  int b = bh >> 4, h = bh & 15;
  const __bf16* Pb = P + (size_t)bh * SEQL * SEQL;
  const __bf16* Vb = V + (size_t)b * SEQL * EMBD + h * HDIM;
  __bf16* Ob = Obf + (size_t)b * SEQL * EMBD + h * HDIM;

  __shared__ __bf16 As[128][32];
  __shared__ __bf16 BsT[64][BPAD];
  int tid = threadIdx.x;
  int wave = tid >> 5;
  int wm = wave >> 1, wn = wave & 1;
  int bm = blockIdx.y * 128;

  v8f acc[2][2] = {};
  for (int k0 = 0; k0 < SEQL; k0 += 32) {
    stage_a_128x32(Pb + (size_t)bm * SEQL + k0, SEQL, As);
    stage_bT_transpose(Vb + (size_t)k0 * EMBD, EMBD, BsT);
    wait_async_lds();
    __syncthreads();
    v16bf a0 = frag_a(&As[wm * 32][0], 32);
    v16bf a1 = frag_a(&As[wm * 32 + 16][0], 32);
    v16bf b0 = frag_bT(&BsT[wn * 32][0]);
    v16bf b1 = frag_bT(&BsT[wn * 32 + 16][0]);
    acc[0][0] = wmma_bf16(a0, b0, acc[0][0]);
    acc[0][1] = wmma_bf16(a0, b1, acc[0][1]);
    acc[1][0] = wmma_bf16(a1, b0, acc[1][0]);
    acc[1][1] = wmma_bf16(a1, b1, acc[1][1]);
    __syncthreads();
  }
  int lane = tid & 31, c = lane & 15, half = lane >> 4;
#pragma unroll
  for (int mi = 0; mi < 2; ++mi)
#pragma unroll
    for (int ni = 0; ni < 2; ++ni)
#pragma unroll
      for (int r = 0; r < 8; ++r) {
        int q = bm + wm * 32 + mi * 16 + half * 8 + r;
        int col = wn * 32 + ni * 16 + c;  // 0..63 head dim
        Ob[(size_t)q * EMBD + col] = (__bf16)acc[mi][ni][r];
      }
}

// y = LayerNorm(x + res) * g + b (fp32 out + bf16 mirror), block per row
__global__ __launch_bounds__(256) void add_ln_kernel(
    const float* __restrict__ X, const float* __restrict__ R,
    const float* __restrict__ g, const float* __restrict__ beta,
    float* __restrict__ Y, __bf16* __restrict__ Ybf) {
  size_t row = blockIdx.x;
  const float* x = X + row * EMBD;
  const float* r = R + row * EMBD;
  __shared__ float red[256];
  float v[4];
  float s = 0.0f;
#pragma unroll
  for (int j = 0; j < 4; ++j) {
    int i = threadIdx.x + j * 256;
    v[j] = x[i] + r[i];
    s += v[j];
  }
  red[threadIdx.x] = s;
  __syncthreads();
  for (int off = 128; off > 0; off >>= 1) {
    if (threadIdx.x < off) red[threadIdx.x] += red[threadIdx.x + off];
    __syncthreads();
  }
  float mu = red[0] * (1.0f / EMBD);
  __syncthreads();
  s = 0.0f;
#pragma unroll
  for (int j = 0; j < 4; ++j) {
    float d = v[j] - mu;
    s += d * d;
  }
  red[threadIdx.x] = s;
  __syncthreads();
  for (int off = 128; off > 0; off >>= 1) {
    if (threadIdx.x < off) red[threadIdx.x] += red[threadIdx.x + off];
    __syncthreads();
  }
  float rstd = rsqrtf(red[0] * (1.0f / EMBD) + 1e-5f);
  __syncthreads();
#pragma unroll
  for (int j = 0; j < 4; ++j) {
    int i = threadIdx.x + j * 256;
    float o = (v[j] - mu) * rstd * g[i] + beta[i];
    Y[row * EMBD + i] = o;
    Ybf[row * EMBD + i] = (__bf16)o;
  }
}

// h[row][e] = emb[tok[row]][e]*sqrt(E) + pe(l, e)  (fp32 + bf16 mirror)
__global__ __launch_bounds__(256) void embed_kernel(
    const int* __restrict__ tok, const float* __restrict__ emb,
    float* __restrict__ H, __bf16* __restrict__ Hbf) {
  int row = blockIdx.x;
  int l = row & (SEQL - 1);
  int t = tok[row];
  const float* e = emb + (size_t)t * EMBD;
  const float LOG1E4_OVER_E = 9.210340371976184f / (float)EMBD;
  for (int i = threadIdx.x; i < EMBD; i += 256) {
    float freq = __expf(-(float)(i & ~1) * LOG1E4_OVER_E);
    float th = (float)l * freq;
    float pe = (i & 1) ? __cosf(th) : __sinf(th);
    float o = e[i] * 32.0f + pe;  // sqrt(1024) = 32
    H[(size_t)row * EMBD + i] = o;
    Hbf[(size_t)row * EMBD + i] = (__bf16)o;
  }
}

__global__ __launch_bounds__(256) void f32_to_bf16_kernel(
    const float* __restrict__ in, __bf16* __restrict__ out, size_t n) {
  size_t i = (size_t)blockIdx.x * 256 + threadIdx.x;
  if (i < n) out[i] = (__bf16)in[i];
}

// ---------------------------------------------------------------------------
// Host orchestration.
// Input order = jax.tree_util.tree_leaves(setup_inputs()) (sorted dict keys):
//   dec layer stride 18: +0..3 cross.{wk,wo,wq,wv}, +4 b1, +5 b2, +6 w1, +7 w2,
//     +8 ln1_b, +9 ln1_g, +10 ln2_b, +11 ln2_g, +12 ln3_b, +13 ln3_g,
//     +14..17 self.{wk,wo,wq,wv}
//   enc base 36, stride 12: +0..3 attn.{wk,wo,wq,wv}, +4 b1, +5 b2, +6 w1,
//     +7 w2, +8 ln1_b, +9 ln1_g, +10 ln2_b, +11 ln2_g
//   60 inp_emb, 61 out_b, 62 out_w, 63 tgt_emb, 64 x, 65 y
// ---------------------------------------------------------------------------
struct AttnW { const __bf16 *wq, *wk, *wv, *wo; };

extern "C" void kernel_launch(void* const* d_in, const int* in_sizes, int n_in,
                              void* d_out, int out_size, void* d_ws, size_t ws_size,
                              hipStream_t stream) {
  if (n_in < 66) return;
  const int M = BATCH * SEQL;  // 2048 rows
  size_t ws_off = 0;
  auto walloc = [&](size_t bytes) -> void* {
    void* p = (void*)((char*)d_ws + ws_off);
    ws_off += (bytes + 255) & ~(size_t)255;
    return p;
  };
  auto cvt = [&](int idx, size_t n) -> const __bf16* {
    __bf16* dst = (__bf16*)walloc(n * sizeof(__bf16));
    f32_to_bf16_kernel<<<dim3((unsigned)((n + 255) / 256)), 256, 0, stream>>>(
        (const float*)d_in[idx], dst, n);
    return dst;
  };
  auto gemm = [&](const __bf16* A, const __bf16* B, const float* bias, float* C,
                  __bf16* Cbf, int m, int n, int k, int epi) {
    dim3 grid(n / 64, m / 128);
    if (epi == 0)
      gemm_kernel<0><<<grid, 256, 0, stream>>>(A, B, nullptr, C, Cbf, m, n, k);
    else if (epi == 1)
      gemm_kernel<1><<<grid, 256, 0, stream>>>(A, B, bias, C, Cbf, m, n, k);
    else
      gemm_kernel<3><<<grid, 256, 0, stream>>>(A, B, bias, C, Cbf, m, n, k);
  };

  // ---- convert weights to bf16 ------------------------------------------
  struct EncW { AttnW attn; const __bf16 *w1, *w2; };
  struct DecW { AttnW self, cross; const __bf16 *w1, *w2; };
  EncW enc[2]; DecW dec[2];
  const size_t EE = (size_t)EMBD * EMBD;
  for (int lyr = 0; lyr < 2; ++lyr) {
    int b = lyr * 18;
    dec[lyr].cross = { cvt(b + 2, EE), cvt(b + 0, EE), cvt(b + 3, EE), cvt(b + 1, EE) };
    dec[lyr].self  = { cvt(b + 16, EE), cvt(b + 14, EE), cvt(b + 17, EE), cvt(b + 15, EE) };
    dec[lyr].w1 = cvt(b + 6, (size_t)EMBD * FFND);
    dec[lyr].w2 = cvt(b + 7, (size_t)FFND * EMBD);
  }
  for (int lyr = 0; lyr < 2; ++lyr) {
    int b = 36 + lyr * 12;
    enc[lyr].attn = { cvt(b + 2, EE), cvt(b + 0, EE), cvt(b + 3, EE), cvt(b + 1, EE) };
    enc[lyr].w1 = cvt(b + 6, (size_t)EMBD * FFND);
    enc[lyr].w2 = cvt(b + 7, (size_t)FFND * EMBD);
  }
  const __bf16* out_w = cvt(62, (size_t)EMBD * VOCAB);

  // ---- activation buffers -----------------------------------------------
  size_t actN = (size_t)M * EMBD;
  float*  enc_h  = (float*)walloc(actN * 4);
  __bf16* enc_bf = (__bf16*)walloc(actN * 2);
  float*  dec_t  = (float*)walloc(actN * 4);
  __bf16* dec_bf = (__bf16*)walloc(actN * 2);
  __bf16* qbf    = (__bf16*)walloc(actN * 2);
  __bf16* kbf    = (__bf16*)walloc(actN * 2);
  __bf16* vbf    = (__bf16*)walloc(actN * 2);
  __bf16* ctxbf  = (__bf16*)walloc(actN * 2);
  float*  mha    = (float*)walloc(actN * 4);
  __bf16* ffmbf  = (__bf16*)walloc((size_t)M * FFND * 2);
  float*  ffo    = (float*)walloc(actN * 4);
  float*  sc     = (float*)walloc((size_t)BATCH * NHEAD * SEQL * SEQL * 4);
  __bf16* probs  = (__bf16*)walloc((size_t)BATCH * NHEAD * SEQL * SEQL * 2);
  (void)ws_size; (void)in_sizes; (void)out_size;

  auto attention = [&](const __bf16* qsrc, const __bf16* kvsrc, const AttnW& w,
                       float* out, int causal) {
    gemm(qsrc, w.wq, nullptr, nullptr, qbf, M, EMBD, EMBD, 0);
    gemm(kvsrc, w.wk, nullptr, nullptr, kbf, M, EMBD, EMBD, 0);
    gemm(kvsrc, w.wv, nullptr, nullptr, vbf, M, EMBD, EMBD, 0);
    attn_scores_kernel<<<dim3(SEQL / 64, SEQL / 128, BATCH * NHEAD), 256, 0,
                         stream>>>(qbf, kbf, sc, causal);
    softmax_kernel<<<dim3(BATCH * NHEAD * SEQL), 256, 0, stream>>>(sc, probs);
    attn_pv_kernel<<<dim3(1, SEQL / 128, BATCH * NHEAD), 256, 0, stream>>>(
        probs, vbf, ctxbf);
    gemm(ctxbf, w.wo, nullptr, out, nullptr, M, EMBD, EMBD, 0);
  };
  auto ln = [&](const float* x, const float* res, int gidx, int bidx, float* y,
                __bf16* ybf) {
    add_ln_kernel<<<dim3(M), 256, 0, stream>>>(
        x, res, (const float*)d_in[gidx], (const float*)d_in[bidx], y, ybf);
  };
  auto ffn = [&](const __bf16* xbf, const __bf16* w1, const __bf16* w2,
                 int b1idx, int b2idx) {
    gemm(xbf, w1, (const float*)d_in[b1idx], nullptr, ffmbf, M, FFND, EMBD, 3);
    gemm(ffmbf, w2, (const float*)d_in[b2idx], ffo, nullptr, M, EMBD, FFND, 1);
  };

  // ---- encoder -----------------------------------------------------------
  embed_kernel<<<dim3(M), 256, 0, stream>>>((const int*)d_in[64],
                                            (const float*)d_in[60], enc_h, enc_bf);
  for (int lyr = 0; lyr < 2; ++lyr) {
    int b = 36 + lyr * 12;
    attention(enc_bf, enc_bf, enc[lyr].attn, mha, /*causal=*/0);
    ln(enc_h, mha, b + 9, b + 8, enc_h, enc_bf);
    ffn(enc_bf, enc[lyr].w1, enc[lyr].w2, b + 4, b + 5);
    ln(enc_h, ffo, b + 11, b + 10, enc_h, enc_bf);
  }

  // ---- decoder -----------------------------------------------------------
  embed_kernel<<<dim3(M), 256, 0, stream>>>((const int*)d_in[65],
                                            (const float*)d_in[63], dec_t, dec_bf);
  for (int lyr = 0; lyr < 2; ++lyr) {
    int b = lyr * 18;
    attention(dec_bf, dec_bf, dec[lyr].self, mha, /*causal=*/1);
    ln(dec_t, mha, b + 9, b + 8, dec_t, dec_bf);
    attention(dec_bf, enc_bf, dec[lyr].cross, mha, /*causal=*/0);
    ln(dec_t, mha, b + 11, b + 10, dec_t, dec_bf);
    ffn(dec_bf, dec[lyr].w1, dec[lyr].w2, b + 4, b + 5);
    ln(dec_t, ffo, b + 13, b + 12, dec_t, dec_bf);
  }

  // ---- output projection -------------------------------------------------
  gemm(dec_bf, out_w, (const float*)d_in[61], (float*)d_out, nullptr,
       M, VOCAB, EMBD, 1);
}
